// GraphTransformer_27187142984128
// MI455X (gfx1250) — matrix operand order repaired and checked by reading, not compiled, not run on hardware
//
#include <hip/hip_runtime.h>
#include <math.h>

// ---- problem constants (match reference) ----
#define N_NODES 64000
#define N_EDGES 256000
#define N_GRAPH 2000
#define DIM     300
#define HEADS   4
#define HDIM    75          // DIM / HEADS
#define LAYERS  5
#define EDIM    32
#define FDIM    256
#define HID     1200        // 4*DIM

typedef __attribute__((ext_vector_type(16))) _Float16 v16h;
typedef __attribute__((ext_vector_type(8)))  _Float16 v8h;
typedef __attribute__((ext_vector_type(4)))  _Float16 v4h;
typedef __attribute__((ext_vector_type(8)))  float    v8f;

#define GF_BIAS 1
#define GF_RELU 2
#define GF_ACC  4

// GEMM tile config: 256 threads = 8 waves. Block tile 128x64, K-step 32.
// Each wave owns one 16-row strip and all four 16-col tiles (4 WMMAs/K-step,
// A-fragment reused 4x).
#define BM 128
#define BN 64
#define BK 32

// ---------------------------------------------------------------------------
// LDS-staged f16-operand / f32-accumulate WMMA GEMM.
//   C = act((acc ? C : 0) + A@W + bias)
// A: [M,K] f32 row-major, W: [K,Nc] f32 row-major, C: [M,Nc] f32.
// Staging: float4 global loads (addresses clamped in-bounds, tails zeroed by
// selects), loads batched before LDS writes. Fragments are contiguous LDS
// reads in k = 16*half + s order, the same order for A and B.
// ---------------------------------------------------------------------------
__global__ __launch_bounds__(256) void gemm_wmma_f16(
    const float* __restrict__ A, const float* __restrict__ W,
    const float* __restrict__ bias, float* __restrict__ C,
    int M, int K, int Nc, int flags)
{
  __shared__ __align__(64) _Float16 lds_a[BM * BK];   // [BM][BK] row-major
  __shared__ __align__(64) _Float16 lds_bt[BN * BK];  // [BN][BK] = B transposed

  const int tid  = threadIdx.x;
  const int wave = tid >> 5;
  const int lane = tid & 31;
  const int half = lane >> 4;          // 0: lanes 0-15, 1: lanes 16-31
  const int r    = lane & 15;

  const int mBase = blockIdx.y * BM;
  const int nBase = blockIdx.x * BN;

  const size_t amax_lin = (size_t)M * (size_t)K - 4;    // clamp for float4 reads
  const size_t wmax_lin = (size_t)K * (size_t)Nc - 4;

  v8f acc0 = {}, acc1 = {}, acc2 = {}, acc3 = {};

  const int ksteps = (K + BK - 1) / BK;
  for (int ks = 0; ks < ksteps; ++ks) {
    const int kbase = ks * BK;

    // ---- phase 1: batched global float4 loads (A: 4/thread, B: 2/thread) ----
    float4 aval[4];
#pragma unroll
    for (int i = 0; i < 4; ++i) {
      int idx4 = tid + i * 256;              // 1024 float4 = 128x32 tile
      int rr = idx4 >> 3;                    // / (BK/4)
      int kk = (idx4 & 7) << 2;              // % (BK/4) * 4
      int grow = mBase + rr; if (grow >= M) grow = M - 1;
      size_t lin = (size_t)grow * K + (size_t)(kbase + kk);
      if (lin > amax_lin) lin = amax_lin;
      aval[i] = *(const float4*)(A + lin);
    }
    float4 bval[2];
#pragma unroll
    for (int i = 0; i < 2; ++i) {
      int idx4 = tid + i * 256;              // 512 float4 = 32x64 tile
      int kk = idx4 >> 4;                    // / (BN/4)
      int cc = (idx4 & 15) << 2;             // % (BN/4) * 4
      int gk = kbase + kk; if (gk >= K) gk = K - 1;
      size_t lin = (size_t)gk * Nc + (size_t)(nBase + cc);
      if (lin > wmax_lin) lin = wmax_lin;
      bval[i] = *(const float4*)(W + lin);
    }

    // ---- phase 2: convert + zero-fill tails, write LDS ----
#pragma unroll
    for (int i = 0; i < 4; ++i) {
      int idx4 = tid + i * 256;
      int rr = idx4 >> 3;
      int kk = (idx4 & 7) << 2;
      int gk = kbase + kk;
      v4h hv;
      hv[0] = (_Float16)((gk + 0 < K) ? aval[i].x : 0.0f);
      hv[1] = (_Float16)((gk + 1 < K) ? aval[i].y : 0.0f);
      hv[2] = (_Float16)((gk + 2 < K) ? aval[i].z : 0.0f);
      hv[3] = (_Float16)((gk + 3 < K) ? aval[i].w : 0.0f);
      *(v4h*)(lds_a + rr * BK + kk) = hv;    // 8B packed store
    }
#pragma unroll
    for (int i = 0; i < 2; ++i) {
      int idx4 = tid + i * 256;
      int kk = idx4 >> 4;
      int cc = (idx4 & 15) << 2;
      int gk = kbase + kk;
      float bf[4] = {bval[i].x, bval[i].y, bval[i].z, bval[i].w};
#pragma unroll
      for (int j = 0; j < 4; ++j) {
        int gn = nBase + cc + j;
        float vv = (gk < K && gn < Nc) ? bf[j] : 0.0f;
        lds_bt[(cc + j) * BK + kk] = (_Float16)vv;   // transposed
      }
    }
    __syncthreads();

    // ---- fragments + 4 WMMAs ----
    // A-frag (16x32 f16): lane half h holds 16 consecutive K at 16h..16h+15
    const int arow  = wave * 16 + r;
    v16h afrag = *(const v16h*)(lds_a + arow * BK + (half << 4));
    // B-frag (32x16 f16): 16 consecutive K at column c (transposed LDS)
    const int koff = half << 4;
    v16h b0 = *(const v16h*)(lds_bt + (r)      * BK + koff);
    v16h b1 = *(const v16h*)(lds_bt + (16 + r) * BK + koff);
    v16h b2 = *(const v16h*)(lds_bt + (32 + r) * BK + koff);
    v16h b3 = *(const v16h*)(lds_bt + (48 + r) * BK + koff);

    acc0 = __builtin_amdgcn_wmma_f32_16x16x32_f16(false, afrag, false, b0, (short)0, acc0, false, false);
    acc1 = __builtin_amdgcn_wmma_f32_16x16x32_f16(false, afrag, false, b1, (short)0, acc1, false, false);
    acc2 = __builtin_amdgcn_wmma_f32_16x16x32_f16(false, afrag, false, b2, (short)0, acc2, false, false);
    acc3 = __builtin_amdgcn_wmma_f32_16x16x32_f16(false, afrag, false, b3, (short)0, acc3, false, false);
    __syncthreads();
  }

  // ---- store: C/D layout row = 16*tile + half*8 + vr, col = lane&15 ----
  const int crow0 = mBase + wave * 16 + (half << 3);
#pragma unroll
  for (int t = 0; t < 4; ++t) {
    v8f acc = (t == 0) ? acc0 : (t == 1) ? acc1 : (t == 2) ? acc2 : acc3;
    int ccol = nBase + t * 16 + r;
    if (ccol < Nc) {
      float bv = (flags & GF_BIAS) ? bias[ccol] : 0.0f;
#pragma unroll
      for (int vr = 0; vr < 8; ++vr) {
        int crow = crow0 + vr;
        if (crow < M) {
          size_t off = (size_t)crow * Nc + ccol;
          float val = acc[vr] + bv;
          if (flags & GF_ACC)  val += C[off];
          if (flags & GF_RELU) val = fmaxf(val, 0.0f);
          C[off] = val;
        }
      }
    }
  }
}

// ---------------------------------------------------------------------------
// Elementwise / scatter helpers
// ---------------------------------------------------------------------------
__global__ void fill_f32(float* p, float v, long n) {
  long i = (long)blockIdx.x * blockDim.x + threadIdx.x;
  if (i < n) p[i] = v;
}

__global__ void embed_nodes_kernel(const int* __restrict__ x,
                                   const float* __restrict__ e1,
                                   const float* __restrict__ e2,
                                   float* __restrict__ h, long total) {
  long i = (long)blockIdx.x * blockDim.x + threadIdx.x;
  if (i >= total) return;
  int n = (int)(i / DIM), d = (int)(i % DIM);
  h[i] = e1[(size_t)x[2 * n] * DIM + d] + e2[(size_t)x[2 * n + 1] * DIM + d];
}

__global__ void embed_edges_kernel(const int* __restrict__ ea,
                                   const float* __restrict__ et,
                                   const float* __restrict__ ed,
                                   float* __restrict__ tpre, long total) {
  long i = (long)blockIdx.x * blockDim.x + threadIdx.x;
  if (i >= total) return;
  int e = (int)(i / EDIM), c = (int)(i % EDIM);
  tpre[i] = et[(size_t)ea[2 * e] * EDIM + c] + ed[(size_t)ea[2 * e + 1] * EDIM + c];
}

__device__ inline void atomicMaxFloat(float* addr, float val) {
  int iv = __float_as_int(val);
  if (iv >= 0) atomicMax((int*)addr, iv);
  else         atomicMin((unsigned int*)addr, (unsigned int)iv);
}

// alpha[e,h] = scale * <q[dst], k[src]+e_edge>, plus segment-max into amax[dst,h]
__global__ void alpha_kernel(const int* __restrict__ edge_index,
                             const float* __restrict__ q, const float* __restrict__ k,
                             const float* __restrict__ eed,
                             float* __restrict__ alpha, float* __restrict__ amax,
                             float scale, long total) {
  long i = (long)blockIdx.x * blockDim.x + threadIdx.x;
  if (i >= total) return;
  int e = (int)(i / HEADS), hh = (int)(i % HEADS);
  int s  = edge_index[e];
  int dn = edge_index[N_EDGES + e];
  const float* qp = q   + (size_t)dn * DIM + hh * HDIM;
  const float* kp = k   + (size_t)s  * DIM + hh * HDIM;
  const float* ep = eed + (size_t)e  * DIM + hh * HDIM;
  float sum = 0.f;
  for (int c = 0; c < HDIM; ++c) sum += qp[c] * (kp[c] + ep[c]);
  sum *= scale;
  alpha[i] = sum;
  atomicMaxFloat(&amax[(size_t)dn * HEADS + hh], sum);
}

// alpha <- exp(alpha - amax[dst]);  den[dst] += alpha
__global__ void softmax_num_kernel(const int* __restrict__ edge_index,
                                   float* __restrict__ alpha,
                                   const float* __restrict__ amax,
                                   float* __restrict__ den, long total) {
  long i = (long)blockIdx.x * blockDim.x + threadIdx.x;
  if (i >= total) return;
  int e = (int)(i / HEADS), hh = (int)(i % HEADS);
  int dn = edge_index[N_EDGES + e];
  float v = expf(alpha[i] - amax[(size_t)dn * HEADS + hh]);
  alpha[i] = v;
  atomicAdd(&den[(size_t)dn * HEADS + hh], v);
}

// agg[dst] += (alpha/den[dst]) * (v[src] + e_edge)
__global__ void scatter_msg_kernel(const int* __restrict__ edge_index,
                                   const float* __restrict__ alpha,
                                   const float* __restrict__ den,
                                   const float* __restrict__ v,
                                   const float* __restrict__ eed,
                                   float* __restrict__ agg, long total) {
  long i = (long)blockIdx.x * blockDim.x + threadIdx.x;
  if (i >= total) return;
  int e = (int)(i / HEADS), hh = (int)(i % HEADS);
  int s  = edge_index[e];
  int dn = edge_index[N_EDGES + e];
  float attn = alpha[i] / den[(size_t)dn * HEADS + hh];
  const float* vp = v   + (size_t)s * DIM + hh * HDIM;
  const float* ep = eed + (size_t)e * DIM + hh * HDIM;
  float* ap = agg + (size_t)dn * DIM + hh * HDIM;
  for (int c = 0; c < HDIM; ++c) atomicAdd(&ap[c], attn * (vp[c] + ep[c]));
}

// out = LayerNorm(a + b) * g + be   (one thread per row)
__global__ void ln_residual_kernel(const float* __restrict__ a, const float* __restrict__ b,
                                   const float* __restrict__ g, const float* __restrict__ be,
                                   float* __restrict__ out, int M) {
  int row = blockIdx.x * blockDim.x + threadIdx.x;
  if (row >= M) return;
  const float* ap = a + (size_t)row * DIM;
  const float* bp = b + (size_t)row * DIM;
  float m = 0.f;
  for (int d = 0; d < DIM; ++d) m += ap[d] + bp[d];
  m /= (float)DIM;
  float var = 0.f;
  for (int d = 0; d < DIM; ++d) { float t = ap[d] + bp[d] - m; var += t * t; }
  var /= (float)DIM;
  float rs = rsqrtf(var + 1e-5f);
  float* op = out + (size_t)row * DIM;
  for (int d = 0; d < DIM; ++d) op[d] = (ap[d] + bp[d] - m) * rs * g[d] + be[d];
}

__global__ void count_kernel(const int* __restrict__ batch, float* __restrict__ cnt, int n) {
  int i = blockIdx.x * blockDim.x + threadIdx.x;
  if (i < n) atomicAdd(&cnt[batch[i]], 1.0f);
}

__global__ void pool_scatter_kernel(const int* __restrict__ batch,
                                    const float* __restrict__ h,
                                    float* __restrict__ gsum, long total) {
  long i = (long)blockIdx.x * blockDim.x + threadIdx.x;
  if (i >= total) return;
  int n = (int)(i / DIM), d = (int)(i % DIM);
  atomicAdd(&gsum[(size_t)batch[n] * DIM + d], h[i]);
}

__global__ void pool_norm_kernel(float* __restrict__ gsum, const float* __restrict__ cnt, long total) {
  long i = (long)blockIdx.x * blockDim.x + threadIdx.x;
  if (i >= total) return;
  gsum[i] /= fmaxf(cnt[i / DIM], 1.0f);
}

// ---------------------------------------------------------------------------
// Host-side orchestration
// ---------------------------------------------------------------------------
static inline void launch_gemm(const float* A, const float* W, const float* bias, float* C,
                               int M, int K, int Nc, int flags, hipStream_t s) {
  dim3 grid((Nc + BN - 1) / BN, (M + BM - 1) / BM);
  gemm_wmma_f16<<<grid, 256, 0, s>>>(A, W, bias, C, M, K, Nc, flags);
}
static inline dim3 g1(long n) { return dim3((unsigned)((n + 255) / 256)); }

extern "C" void kernel_launch(void* const* d_in, const int* in_sizes, int n_in,
                              void* d_out, int out_size, void* d_ws, size_t ws_size,
                              hipStream_t stream) {
  (void)in_sizes; (void)n_in; (void)out_size; (void)ws_size;
  const int*   x           = (const int*)d_in[0];
  const int*   edge_index  = (const int*)d_in[1];
  const int*   edge_attr   = (const int*)d_in[2];
  const int*   batch       = (const int*)d_in[3];
  const float* x_emb1      = (const float*)d_in[4];
  const float* x_emb2      = (const float*)d_in[5];
  const float* e_emb_t     = (const float*)d_in[6];
  const float* e_emb_d     = (const float*)d_in[7];
  const float* edge_proj_w = (const float*)d_in[8];
  const float* edge_proj_b = (const float*)d_in[9];
  const float* q_w  = (const float*)d_in[10]; const float* q_b  = (const float*)d_in[11];
  const float* k_w  = (const float*)d_in[12]; const float* k_b  = (const float*)d_in[13];
  const float* v_w  = (const float*)d_in[14]; const float* v_b  = (const float*)d_in[15];
  const float* e_w  = (const float*)d_in[16];
  const float* skip_w = (const float*)d_in[17]; const float* skip_b = (const float*)d_in[18];
  const float* ln1_g = (const float*)d_in[19]; const float* ln1_b = (const float*)d_in[20];
  const float* ln2_g = (const float*)d_in[21]; const float* ln2_b = (const float*)d_in[22];
  const float* ffn_w1 = (const float*)d_in[23]; const float* ffn_b1 = (const float*)d_in[24];
  const float* ffn_w2 = (const float*)d_in[25]; const float* ffn_b2 = (const float*)d_in[26];
  const float* feat_w = (const float*)d_in[27]; const float* feat_b = (const float*)d_in[28];
  const float* out_w1 = (const float*)d_in[29]; const float* out_b1 = (const float*)d_in[30];
  const float* out_w2 = (const float*)d_in[31]; const float* out_b2 = (const float*)d_in[32];

  // ---- workspace layout (floats) ----
  float* ws = (float*)d_ws;
  const size_t SZ_ND = (size_t)N_NODES * DIM;            // 19.2M
  float* h    = ws;
  float* q    = ws + 1 * SZ_ND;
  float* k    = ws + 2 * SZ_ND;
  float* v    = ws + 3 * SZ_ND;
  float* agg  = ws + 4 * SZ_ND;
  float* big  = ws + 5 * SZ_ND;                          // E*DIM == N*HID == 76.8M
  float* ef    = big   + (size_t)N_EDGES * DIM;          // E*EDIM
  float* alpha = ef    + (size_t)N_EDGES * EDIM;         // E*H
  float* amax  = alpha + (size_t)N_EDGES * HEADS;        // N*H
  float* den   = amax  + (size_t)N_NODES * HEADS;        // N*H
  float* gsum  = den   + (size_t)N_NODES * HEADS;        // G*DIM
  float* cnt   = gsum  + (size_t)N_GRAPH * DIM;          // G
  float* t1    = cnt   + (size_t)N_GRAPH;                // G*FDIM

  float* g_out  = (float*)d_out;                         // [G, FDIM]
  float* o_out  = g_out + (size_t)N_GRAPH * FDIM;        // [G, FDIM/2]

  const float scale = 1.0f / sqrtf((float)HDIM);
  const long ND = (long)N_NODES * DIM;
  const long EH = (long)N_EDGES * HEADS;

  // ---- embeddings ----
  embed_nodes_kernel<<<g1(ND), 256, 0, stream>>>(x, x_emb1, x_emb2, h, ND);
  embed_edges_kernel<<<g1((long)N_EDGES * EDIM), 256, 0, stream>>>(edge_attr, e_emb_t, e_emb_d,
                                                                   big, (long)N_EDGES * EDIM);
  launch_gemm(big, edge_proj_w, edge_proj_b, ef, N_EDGES, EDIM, EDIM, GF_BIAS, stream);

  // ---- transformer layers ----
  for (int l = 0; l < LAYERS; ++l) {
    const float* qw = q_w + (size_t)l * DIM * DIM;  const float* qb = q_b + (size_t)l * DIM;
    const float* kw = k_w + (size_t)l * DIM * DIM;  const float* kb = k_b + (size_t)l * DIM;
    const float* vw = v_w + (size_t)l * DIM * DIM;  const float* vb = v_b + (size_t)l * DIM;
    const float* ew = e_w + (size_t)l * EDIM * DIM;
    const float* sw = skip_w + (size_t)l * DIM * DIM; const float* sb = skip_b + (size_t)l * DIM;

    launch_gemm(h,  qw, qb, q,   N_NODES, DIM, DIM, GF_BIAS, stream);
    launch_gemm(h,  kw, kb, k,   N_NODES, DIM, DIM, GF_BIAS, stream);
    launch_gemm(h,  vw, vb, v,   N_NODES, DIM, DIM, GF_BIAS, stream);
    launch_gemm(ef, ew, nullptr, big, N_EDGES, EDIM, DIM, 0, stream);   // per-edge features e

    fill_f32<<<g1((long)N_NODES * HEADS), 256, 0, stream>>>(amax, -INFINITY, (long)N_NODES * HEADS);
    fill_f32<<<g1((long)N_NODES * HEADS), 256, 0, stream>>>(den, 0.0f, (long)N_NODES * HEADS);
    fill_f32<<<g1(ND), 256, 0, stream>>>(agg, 0.0f, ND);

    alpha_kernel<<<g1(EH), 256, 0, stream>>>(edge_index, q, k, big, alpha, amax, scale, EH);
    softmax_num_kernel<<<g1(EH), 256, 0, stream>>>(edge_index, alpha, amax, den, EH);
    scatter_msg_kernel<<<g1(EH), 256, 0, stream>>>(edge_index, alpha, den, v, big, agg, EH);

    // conv = agg + h @ skip_w + skip_b  (accumulate into agg)
    launch_gemm(h, sw, sb, agg, N_NODES, DIM, DIM, GF_BIAS | GF_ACC, stream);
    // h = LN(h + conv)
    ln_residual_kernel<<<g1(N_NODES), 256, 0, stream>>>(h, agg, ln1_g + (size_t)l * DIM,
                                                        ln1_b + (size_t)l * DIM, h, N_NODES);
    // FFN
    launch_gemm(h, ffn_w1 + (size_t)l * DIM * HID, ffn_b1 + (size_t)l * HID,
                big, N_NODES, DIM, HID, GF_BIAS | GF_RELU, stream);
    launch_gemm(big, ffn_w2 + (size_t)l * HID * DIM, ffn_b2 + (size_t)l * DIM,
                agg, N_NODES, HID, DIM, GF_BIAS, stream);
    ln_residual_kernel<<<g1(N_NODES), 256, 0, stream>>>(h, agg, ln2_g + (size_t)l * DIM,
                                                        ln2_b + (size_t)l * DIM, h, N_NODES);
  }

  // ---- mean pooling over graphs ----
  fill_f32<<<g1((long)N_GRAPH * DIM), 256, 0, stream>>>(gsum, 0.0f, (long)N_GRAPH * DIM);
  fill_f32<<<g1((long)N_GRAPH), 256, 0, stream>>>(cnt, 0.0f, (long)N_GRAPH);
  count_kernel<<<g1((long)N_NODES), 256, 0, stream>>>(batch, cnt, N_NODES);
  pool_scatter_kernel<<<g1(ND), 256, 0, stream>>>(batch, h, gsum, ND);
  pool_norm_kernel<<<g1((long)N_GRAPH * DIM), 256, 0, stream>>>(gsum, cnt, (long)N_GRAPH * DIM);

  // ---- heads ----
  launch_gemm(gsum, feat_w, feat_b, g_out, N_GRAPH, DIM, FDIM, GF_BIAS, stream);          // g
  launch_gemm(g_out, out_w1, out_b1, t1, N_GRAPH, FDIM, FDIM, GF_BIAS | GF_RELU, stream); // relu MLP
  launch_gemm(t1, out_w2, out_b2, o_out, N_GRAPH, FDIM, FDIM / 2, GF_BIAS, stream);       // out
}
